// StackedWaveNet_61125974556799
// MI455X (gfx1250) — matrix operand order repaired and checked by reading, not compiled, not run on hardware
//
#include <hip/hip_runtime.h>
#include <math.h>

typedef __bf16 bf16;
typedef bf16  v16bf __attribute__((ext_vector_type(16)));
typedef bf16  v8bf  __attribute__((ext_vector_type(8)));
typedef float v8f   __attribute__((ext_vector_type(8)));
typedef int   v4i_vs __attribute__((vector_size(16)));
typedef __attribute__((address_space(1))) v4i_vs* as1_v4i_ptr;
typedef __attribute__((address_space(3))) v4i_vs* as3_v4i_ptr;

#define WN_B 4
#define WN_T 16384
#define WN_L 30
#define WN_R 64
#define WN_S 128

// per-layer packed weight block (bf16 elements): [pf 8192][pg 8192][pr 4096][ps 8192]
#define OFF_F 0
#define OFF_G 8192
#define OFF_R 16384
#define OFF_S 20480
#define PACK_PER_LAYER 28672

#if __has_builtin(__builtin_amdgcn_global_load_async_to_lds_b128)
#define WN_HAVE_ASYNC_LDS 1
#else
#define WN_HAVE_ASYNC_LDS 0
#endif

static __device__ __forceinline__ v8f wmma_bf16(v16bf a, v16bf b, v8f c) {
    // D = A(16x32) * B(32x16) + C, f32 accumulate
    return __builtin_amdgcn_wmma_f32_16x16x32_bf16(false, a, false, b, (short)0, c,
                                                   false, false);
}

static __device__ __forceinline__ v8f v8f_zero() {
    v8f z;
#pragma unroll
    for (int i = 0; i < 8; ++i) z[i] = 0.0f;
    return z;
}

static __device__ __forceinline__ v16bf v16bf_zero() {
    v16bf z;
#pragma unroll
    for (int i = 0; i < 16; ++i) z[i] = (bf16)0.0f;
    return z;
}

// Cooperative copy of `nvec` 16-byte vectors global -> LDS, via the CDNA5
// async global->LDS path when available (ASYNCcnt), else plain copy.
static __device__ __forceinline__ void stage_to_lds(bf16* lds_dst, const bf16* gsrc,
                                                    int nvec) {
    for (int i = threadIdx.x; i < nvec; i += 256) {
#if WN_HAVE_ASYNC_LDS
        __builtin_amdgcn_global_load_async_to_lds_b128(
            (as1_v4i_ptr)(void*)(gsrc + (size_t)i * 8),
            (as3_v4i_ptr)(void*)(lds_dst + (size_t)i * 8), 0, 0);
#else
        *(v8bf*)(lds_dst + (size_t)i * 8) = *(const v8bf*)(gsrc + (size_t)i * 8);
#endif
    }
#if WN_HAVE_ASYNC_LDS
    asm volatile("s_wait_asynccnt 0" ::: "memory");
#endif
    __syncthreads();
}

// ---------------------------------------------------------------------------
// Weight packing: rearrange fp32 weights into WMMA A-fragment order (bf16).
// A k order used here (contiguous, B matches): lane l holds row m=l&15; element j (0..15) holds
// K(l,j) = j + (l>=16 ? 16 : 0).
// ---------------------------------------------------------------------------

// kernel-2 conv weights W[L][O=64][C=64][2] -> per-layer [mt=4][ks=4][lane=32][j=16]
// ks 0,1 -> tap 0 (x[t-d]) channels 0-31 / 32-63 ; ks 2,3 -> tap 1 (x[t]).
__global__ void wn_pack_k2_kernel(bf16* __restrict__ dst, const float* __restrict__ W,
                                  int nlayer, int dstStride, int dstOff) {
    int idx = blockIdx.x * blockDim.x + threadIdx.x;
    int total = nlayer * 8192;
    if (idx >= total) return;
    int j     = idx & 15;
    int lane  = (idx >> 4) & 31;
    int ks    = (idx >> 9) & 3;
    int mt    = (idx >> 11) & 3;
    int layer = idx >> 13;
    int m  = mt * 16 + (lane & 15);
    int kk = j + ((lane >= 16) ? 16 : 0);
    int c  = (ks & 1) * 32 + kk;
    int tap = ks >> 1;
    float v = W[(((size_t)layer * 64 + m) * 64 + c) * 2 + tap];
    dst[(size_t)layer * dstStride + dstOff + (idx & 8191)] = (bf16)v;
}

// 1x1 conv weights W[L][O][C] -> per-layer [mt=O/16][ks=C/32][lane][j]
__global__ void wn_pack_1x1_kernel(bf16* __restrict__ dst, const float* __restrict__ W,
                                   int nlayer, int O, int C, int dstStride, int dstOff) {
    int mt_n = O / 16, ks_n = C / 32;
    int per_layer = mt_n * ks_n * 512;
    int idx = blockIdx.x * blockDim.x + threadIdx.x;
    int total = nlayer * per_layer;
    if (idx >= total) return;
    int within = idx % per_layer;
    int layer  = idx / per_layer;
    int j    = within & 15;
    int lane = (within >> 4) & 31;
    int rem  = within >> 9;
    int ks   = rem % ks_n;
    int mt   = rem / ks_n;
    int m  = mt * 16 + (lane & 15);
    int kk = j + ((lane >= 16) ? 16 : 0);
    int c  = ks * 32 + kk;
    dst[(size_t)layer * dstStride + dstOff + within] = (bf16)W[((size_t)layer * O + m) * C + c];
}

// ---------------------------------------------------------------------------
// Initial causal conv (1 -> 64 channels, k=2, dilation 1).
// h stored time-major: h[(b*T+t)*64 + o], fp32 + bf16 mirror.
// ---------------------------------------------------------------------------
__global__ void wn_init_kernel(const float* __restrict__ x,
                               const float* __restrict__ w_causal,
                               const float* __restrict__ b_causal,
                               float* __restrict__ hf, bf16* __restrict__ hb) {
    int idx = blockIdx.x * blockDim.x + threadIdx.x;  // grid sized exactly
    int o = idx & 63;
    int t = (idx >> 6) & (WN_T - 1);
    int b = idx >> 20;  // T*64 = 2^20
    float xc = x[(size_t)b * WN_T + t];
    float xp = (t > 0) ? x[(size_t)b * WN_T + t - 1] : 0.0f;
    float v = w_causal[o * 2 + 0] * xp + w_causal[o * 2 + 1] * xc + b_causal[o];
    hf[idx] = v;
    hb[idx] = (bf16)v;
}

// ---------------------------------------------------------------------------
// One dilated gated residual layer. 1 wave == 16 time steps of one batch.
// Layer weights staged once per workgroup into LDS (async global->LDS).
// ---------------------------------------------------------------------------
__global__ __launch_bounds__(256) void wn_layer_kernel(
    const bf16* __restrict__ h_in, bf16* __restrict__ h_out,
    float* __restrict__ hf, float* __restrict__ skip,
    const bf16* __restrict__ pw,
    const float* __restrict__ bf_, const float* __restrict__ bg_,
    const float* __restrict__ br_, const float* __restrict__ bs_,
    int dil, int first) {
    __shared__ __align__(32) bf16 lds_w[PACK_PER_LAYER];  // [pf|pg|pr|ps]
    __shared__ __align__(32) bf16 lds_out[8][16][80];     // per-wave 16 x 64ch (+pad)

    // --- stage this layer's packed weights into LDS (shared by 8 waves) ---
    stage_to_lds(lds_w, pw, PACK_PER_LAYER / 8);

    const int lane = threadIdx.x & 31;
    const int wv   = threadIdx.x >> 5;
    const int tile = blockIdx.x * 8 + wv;
    const int tilesPerB = WN_T / 16;
    const int b  = tile / tilesPerB;
    const int t0 = (tile % tilesPerB) * 16;
    const int n  = lane & 15;
    const int hi = lane >> 4;
    const int tc = t0 + n;
    const size_t rowc = ((size_t)b * WN_T + tc) * 64;
    const int lofs = lane * 16;

    // --- B fragments for the gate GEMMs (K = 2 taps x 64 ch = 4 k-steps) ---
    v16bf Bf[4];
#pragma unroll
    for (int ks = 0; ks < 4; ++ks) {
        int tap = ks >> 1;                   // 0: x[t-d], 1: x[t]
        int t = tc - (tap == 0 ? dil : 0);
        int kc = (ks & 1) * 32 + hi * 16;
        if (t >= 0)
            Bf[ks] = *(const v16bf*)(h_in + ((size_t)b * WN_T + t) * 64 + kc);
        else
            Bf[ks] = v16bf_zero();
    }

    // --- F and G: 4 M-tiles x 4 K-steps each, A-fragments from LDS ---
    v8f F[4], G[4];
#pragma unroll
    for (int mt = 0; mt < 4; ++mt) { F[mt] = v8f_zero(); G[mt] = v8f_zero(); }
#pragma unroll
    for (int mt = 0; mt < 4; ++mt) {
#pragma unroll
        for (int ks = 0; ks < 4; ++ks) {
            v16bf af = *(const v16bf*)&lds_w[OFF_F + (mt * 4 + ks) * 512 + lofs];
            F[mt] = wmma_bf16(af, Bf[ks], F[mt]);
            v16bf ag = *(const v16bf*)&lds_w[OFF_G + (mt * 4 + ks) * 512 + lofs];
            G[mt] = wmma_bf16(ag, Bf[ks], G[mt]);
        }
    }

    // --- out = tanh(F) * sigmoid(G); transpose D-layout -> B-layout via LDS ---
#pragma unroll
    for (int mt = 0; mt < 4; ++mt) {
        int cb = mt * 16 + hi * 8;
        v8bf ob;
#pragma unroll
        for (int vi = 0; vi < 8; ++vi) {
            float fv = F[mt][vi] + bf_[cb + vi];
            float gv = G[mt][vi] + bg_[cb + vi];
            float o = tanhf(fv) * (1.0f / (1.0f + __expf(-gv)));
            ob[vi] = (bf16)o;
        }
        *(v8bf*)&lds_out[wv][n][cb] = ob;
    }
    __syncthreads();

    v16bf Ob[2];
#pragma unroll
    for (int ks = 0; ks < 2; ++ks)
        Ob[ks] = *(const v16bf*)&lds_out[wv][n][ks * 32 + hi * 16];

    // --- residual: h = Wr@out + br + h  (fp32 in place, bf16 to h_out) ---
#pragma unroll
    for (int mt = 0; mt < 4; ++mt) {
        v8f acc = v8f_zero();
#pragma unroll
        for (int ks = 0; ks < 2; ++ks) {
            v16bf a = *(const v16bf*)&lds_w[OFF_R + (mt * 2 + ks) * 512 + lofs];
            acc = wmma_bf16(a, Ob[ks], acc);
        }
        int cb = mt * 16 + hi * 8;
        float* hp = hf + rowc + cb;
        v8f hold = *(const v8f*)hp;
        v8bf hb;
#pragma unroll
        for (int vi = 0; vi < 8; ++vi) {
            float v = acc[vi] + br_[cb + vi] + hold[vi];
            acc[vi] = v;
            hb[vi] = (bf16)v;
        }
        *(v8f*)hp = acc;
        *(v8bf*)(h_out + rowc + cb) = hb;
    }

    // --- skip accumulation: skip += Ws@out + bs ---
    const size_t rows = ((size_t)b * WN_T + tc) * 128;
#pragma unroll
    for (int mt = 0; mt < 8; ++mt) {
        v8f acc = v8f_zero();
#pragma unroll
        for (int ks = 0; ks < 2; ++ks) {
            v16bf a = *(const v16bf*)&lds_w[OFF_S + (mt * 2 + ks) * 512 + lofs];
            acc = wmma_bf16(a, Ob[ks], acc);
        }
        int cb = mt * 16 + hi * 8;
        float* sp = skip + rows + cb;
        v8f sold;
        if (first) sold = v8f_zero();
        else       sold = *(const v8f*)sp;
#pragma unroll
        for (int vi = 0; vi < 8; ++vi) acc[vi] += bs_[cb + vi] + sold[vi];
        *(v8f*)sp = acc;
    }
}

// ---------------------------------------------------------------------------
// Post 1x1 conv 128->128 with ReLU, in place on skip buffer.
// ---------------------------------------------------------------------------
__global__ __launch_bounds__(256) void wn_post_kernel(
    float* __restrict__ s, const bf16* __restrict__ pw,
    const float* __restrict__ bias) {
    __shared__ __align__(32) bf16 lds_w[16384];  // 128x128 packed fragments

    stage_to_lds(lds_w, pw, 16384 / 8);

    const int lane = threadIdx.x & 31;
    const int wv   = threadIdx.x >> 5;
    const int tile = blockIdx.x * 8 + wv;
    const int tilesPerB = WN_T / 16;
    const int b  = tile / tilesPerB;
    const int t0 = (tile % tilesPerB) * 16;
    const int n  = lane & 15;
    const int hi = lane >> 4;
    const int tc = t0 + n;
    const size_t row = ((size_t)b * WN_T + tc) * 128;
    const int lofs = lane * 16;

    v16bf Bf[4];
#pragma unroll
    for (int ks = 0; ks < 4; ++ks) {
        int kc = ks * 32 + hi * 16;
        const float* p = s + row + kc;
        v16bf v;
#pragma unroll
        for (int i = 0; i < 16; ++i) v[i] = (bf16)p[i];
        Bf[ks] = v;
    }

    v8f R[8];
#pragma unroll
    for (int mt = 0; mt < 8; ++mt) {
        v8f acc = v8f_zero();
#pragma unroll
        for (int ks = 0; ks < 4; ++ks) {
            v16bf a = *(const v16bf*)&lds_w[(mt * 4 + ks) * 512 + lofs];
            acc = wmma_bf16(a, Bf[ks], acc);
        }
        R[mt] = acc;
    }
    // all reads done (in registers) -> safe to overwrite in place
#pragma unroll
    for (int mt = 0; mt < 8; ++mt) {
        int cb = mt * 16 + hi * 8;
        v8f o;
#pragma unroll
        for (int vi = 0; vi < 8; ++vi) {
            float v = R[mt][vi] + bias[cb + vi];
            o[vi] = v > 0.0f ? v : 0.0f;
        }
        *(v8f*)(s + row + cb) = o;
    }
}

// ---------------------------------------------------------------------------
// Final 1x1 conv 128 -> 1.
// ---------------------------------------------------------------------------
__global__ void wn_final_kernel(const float* __restrict__ s,
                                const float* __restrict__ w,
                                const float* __restrict__ b,
                                float* __restrict__ out) {
    int idx = blockIdx.x * blockDim.x + threadIdx.x;  // B*T threads exactly
    const float* p = s + (size_t)idx * 128;
    float acc = b[0];
#pragma unroll 8
    for (int c = 0; c < 128; ++c) acc += w[c] * p[c];
    out[idx] = acc;
}

// ---------------------------------------------------------------------------
extern "C" void kernel_launch(void* const* d_in, const int* in_sizes, int n_in,
                              void* d_out, int out_size, void* d_ws, size_t ws_size,
                              hipStream_t stream) {
    const float* x        = (const float*)d_in[0];
    const float* w_causal = (const float*)d_in[1];
    const float* b_causal = (const float*)d_in[2];
    const float* Wf       = (const float*)d_in[3];
    const float* bfv      = (const float*)d_in[4];
    const float* Wg       = (const float*)d_in[5];
    const float* bgv      = (const float*)d_in[6];
    const float* Wr       = (const float*)d_in[7];
    const float* brv      = (const float*)d_in[8];
    const float* Ws       = (const float*)d_in[9];
    const float* bsv      = (const float*)d_in[10];
    const float* w_post   = (const float*)d_in[11];
    const float* b_post   = (const float*)d_in[12];
    const float* w_o1     = (const float*)d_in[13];
    const float* b_o1     = (const float*)d_in[14];
    const float* w_o2     = (const float*)d_in[15];
    const float* b_o2     = (const float*)d_in[16];

    (void)in_sizes; (void)n_in; (void)out_size; (void)ws_size;

    // ---- workspace carve-up (all 256B aligned) ----
    char* ws = (char*)d_ws;
    size_t off = 0;
    auto carve = [&](size_t bytes) -> void* {
        void* p = ws + off;
        off += (bytes + 255) & ~(size_t)255;
        return p;
    };
    const size_t BT = (size_t)WN_B * WN_T;
    float* hf    = (float*)carve(BT * 64 * sizeof(float));
    bf16*  hb0   = (bf16*) carve(BT * 64 * sizeof(bf16));
    bf16*  hb1   = (bf16*) carve(BT * 64 * sizeof(bf16));
    float* skip  = (float*)carve(BT * 128 * sizeof(float));
    bf16*  packL = (bf16*) carve((size_t)WN_L * PACK_PER_LAYER * sizeof(bf16));
    bf16*  packP = (bf16*) carve(16384 * sizeof(bf16));
    bf16*  packO1= (bf16*) carve(16384 * sizeof(bf16));

    // ---- pack weights into WMMA fragment order (contiguous per-layer block) ----
    {
        int tot = WN_L * 8192;
        wn_pack_k2_kernel<<<(tot + 255) / 256, 256, 0, stream>>>(
            packL, Wf, WN_L, PACK_PER_LAYER, OFF_F);
        wn_pack_k2_kernel<<<(tot + 255) / 256, 256, 0, stream>>>(
            packL, Wg, WN_L, PACK_PER_LAYER, OFF_G);
        int totR = WN_L * 4096;
        wn_pack_1x1_kernel<<<(totR + 255) / 256, 256, 0, stream>>>(
            packL, Wr, WN_L, 64, 64, PACK_PER_LAYER, OFF_R);
        int totS = WN_L * 8192;
        wn_pack_1x1_kernel<<<(totS + 255) / 256, 256, 0, stream>>>(
            packL, Ws, WN_L, 128, 64, PACK_PER_LAYER, OFF_S);
        wn_pack_1x1_kernel<<<(16384 + 255) / 256, 256, 0, stream>>>(
            packP, w_post, 1, 128, 128, 0, 0);
        wn_pack_1x1_kernel<<<(16384 + 255) / 256, 256, 0, stream>>>(
            packO1, w_o1, 1, 128, 128, 0, 0);
    }

    // ---- initial causal conv ----
    {
        int tot = (int)(BT * 64);
        wn_init_kernel<<<tot / 256, 256, 0, stream>>>(x, w_causal, b_causal, hf, hb0);
    }

    // ---- 30 dilated gated residual layers ----
    {
        bf16* hin = hb0;
        bf16* hout = hb1;
        const int nblk = (int)(BT / 16) / 8;  // 512 blocks, 8 waves each
        for (int i = 0; i < WN_L; ++i) {
            int dil = 1 << (i % 10);
            wn_layer_kernel<<<nblk, 256, 0, stream>>>(
                hin, hout, hf, skip,
                packL + (size_t)i * PACK_PER_LAYER,
                bfv + (size_t)i * 64, bgv + (size_t)i * 64,
                brv + (size_t)i * 64, bsv + (size_t)i * 128,
                dil, (i == 0) ? 1 : 0);
            bf16* t = hin; hin = hout; hout = t;
        }
    }

    // ---- post stack: relu(Wpost@skip), relu(Wo1@s), Wo2@s ----
    {
        const int nblk = (int)(BT / 16) / 8;
        wn_post_kernel<<<nblk, 256, 0, stream>>>(skip, packP, b_post);
        wn_post_kernel<<<nblk, 256, 0, stream>>>(skip, packO1, b_o1);
        wn_final_kernel<<<(int)(BT / 256), 256, 0, stream>>>(skip, w_o2, b_o2,
                                                             (float*)d_out);
    }
}
